// MultiHeadAttention_81131932221863
// MI455X (gfx1250) — matrix operand order repaired and checked by reading, not compiled, not run on hardware
//
#include <hip/hip_runtime.h>
#include <hip/hip_bf16.h>

// ---------------------------------------------------------------------------
// MHA forward for MI455X (gfx1250, wave32, WMMA bf16->f32).
// Pipeline: cvt(x) -> pack(W) -> GEMM(qkv) -> RoPE+pack(K,V frags) ->
//           flash-attention (online softmax) -> GEMM(out proj).
// Round 2: 32-row GEMM strips and 32-query attention tiles to raise the
// wmma:load ratio and fill WMMA hazard slots with independent WMMAs.
// ---------------------------------------------------------------------------

typedef __bf16 bf16_t;
typedef __attribute__((ext_vector_type(16))) bf16_t v16bf;
typedef __attribute__((ext_vector_type(8)))  bf16_t v8bf;
typedef __attribute__((ext_vector_type(4)))  bf16_t v4bf;
typedef __attribute__((ext_vector_type(8)))  float  v8f;

#define BB 2
#define TT 2048
#define CC 1024
#define HH 16
#define DD 64

// ---- WMMA fragment helpers (16x16x32 bf16, contiguous k order, not the ISA table order) ---
// A (16x32, row-major source): lane L holds row M=L%16; element e -> K =
// (L/16)*16 + e (contiguous).
__device__ __forceinline__ v16bf load_a_frag_global(const bf16_t* __restrict__ A,
                                                    int lda, int row0, int k0) {
  const int lane = threadIdx.x & 31;
  const bf16_t* p = A + (size_t)(row0 + (lane & 15)) * lda + k0 + ((lane >> 4) << 4);
  v8bf lo = *(const v8bf*)p;          // 16B contiguous
  v8bf hi = *(const v8bf*)(p + 8);    // 16B contiguous
  v16bf r;
#pragma unroll
  for (int i = 0; i < 8; ++i) { r[i] = lo[i]; r[i + 8] = hi[i]; }
  return r;
}

// B (32x16) is pre-packed so each lane's 16 bf16 are contiguous (32B).
__device__ __forceinline__ v16bf load_b_frag(const bf16_t* __restrict__ pack) {
  const int lane = threadIdx.x & 31;
  return *(const v16bf*)(pack + lane * 16);
}

__device__ __forceinline__ v8f wmma_bf16(v16bf a, v16bf b, v8f c) {
  return __builtin_amdgcn_wmma_f32_16x16x32_bf16(false, a, false, b, (short)0, c,
                                                 false, false);
}

// ---------------------------------------------------------------------------
// Kernel 1: f32 -> bf16 row-major convert (4 elems/thread).
// ---------------------------------------------------------------------------
__global__ void __launch_bounds__(256)
f32_to_bf16_v4(const float* __restrict__ in, bf16_t* __restrict__ out) {
  const int i = (blockIdx.x * 256 + threadIdx.x) * 4;
  float4 f = *(const float4*)(in + i);
  v4bf o;
  o[0] = (bf16_t)f.x; o[1] = (bf16_t)f.y; o[2] = (bf16_t)f.z; o[3] = (bf16_t)f.w;
  *(v4bf*)(out + i) = o;
}

// ---------------------------------------------------------------------------
// Kernel 2: pack W (KxN row-major f32) into WMMA B-fragment order (bf16).
// Fragment(nt,kc) is 512 bf16: lane L = ((k%32)/16)*16 + n%16, elem e = k%16.
// ---------------------------------------------------------------------------
__global__ void __launch_bounds__(256)
pack_b_wmma(const float* __restrict__ W, bf16_t* __restrict__ P, int K, int N) {
  const int idx = blockIdx.x * 256 + threadIdx.x;   // over K*N
  const int k = idx / N, n = idx % N;
  const int nt = n >> 4, kc = k >> 5, kk = k & 31;
  const int L = ((kk >> 4) << 4) | (n & 15);
  const int e = kk & 15;
  const int kchunks = K >> 5;
  P[((size_t)(nt * kchunks + kc) << 9) + L * 16 + e] = (bf16_t)W[idx];
}

// ---------------------------------------------------------------------------
// Kernel 3: GEMM  Out(MxN,f32) = A(MxK,bf16 rowmajor) @ Bpack + bias.
// One wave -> 32x64 strip (8 accumulators, B frags shared by both M-tiles).
// ---------------------------------------------------------------------------
__global__ void __launch_bounds__(256)
gemm_bf16_wmma(const bf16_t* __restrict__ A, const bf16_t* __restrict__ Bpack,
               const float* __restrict__ bias, float* __restrict__ Out,
               int M, int N, int K) {
  const int wave = blockIdx.x * 8 + (threadIdx.x >> 5);
  const int ntilesN = N >> 6;
  const int tm = wave / ntilesN;          // 32-row strip index
  const int tn = wave % ntilesN;
  const int lane = threadIdx.x & 31;
  const int kchunks = K >> 5;
  v8f acc[2][4] = {};
  const bf16_t* bp = Bpack + ((size_t)(tn * 4) * kchunks << 9);
  for (int kc = 0; kc < kchunks; ++kc) {
    v16bf a0 = load_a_frag_global(A, K, tm * 32, kc * 32);
    v16bf a1 = load_a_frag_global(A, K, tm * 32 + 16, kc * 32);
#pragma unroll
    for (int j = 0; j < 4; ++j) {
      v16bf b = load_b_frag(bp + ((size_t)(j * kchunks + kc) << 9));
      acc[0][j] = wmma_bf16(a0, b, acc[0][j]);
      acc[1][j] = wmma_bf16(a1, b, acc[1][j]);
    }
  }
  // D layout: lane holds col n=lane%16, rows M = r + 8*(lane/16).
  const int ncol = tn * 64 + (lane & 15);
  float bi[4];
#pragma unroll
  for (int j = 0; j < 4; ++j) bi[j] = bias[ncol + j * 16];
#pragma unroll
  for (int mt = 0; mt < 2; ++mt) {
    const int mrow = tm * 32 + mt * 16 + ((lane >> 4) << 3);
#pragma unroll
    for (int r = 0; r < 8; ++r) {
      float* o = Out + (size_t)(mrow + r) * N + ncol;
#pragma unroll
      for (int j = 0; j < 4; ++j) o[j * 16] = acc[mt][j][r] + bi[j];
    }
  }
}

// ---------------------------------------------------------------------------
// Kernel 4: RoPE on q,k; write Q row-major [B,H,T,D] (A-side), and K,V packed
// into per-head B-fragment order for the attention WMMAs.
//   Kpack[bh]: 128 n-tiles(t/16) x 2 kchunks(d/32) x 512
//   Vpack[bh]:  64 kchunks(t/32) x 4 n-tiles(d/16) x 512
// One thread per (b,h,t,dh<32): handles rope pair (dh, dh+32) + v copy.
// ---------------------------------------------------------------------------
__global__ void __launch_bounds__(256)
rope_pack(const float* __restrict__ qkv, bf16_t* __restrict__ Q,
          bf16_t* __restrict__ Kp, bf16_t* __restrict__ Vp) {
  const int idx = blockIdx.x * 256 + threadIdx.x;       // 2^21 threads
  const int dh = idx & 31;
  const int t  = (idx >> 5) & (TT - 1);
  const int h  = (idx >> 16) & (HH - 1);
  const int b  = idx >> 20;
  const float ang = (float)t * __powf(10000.0f, -(float)dh * (1.0f / 32.0f));
  float s, c;
  __sincosf(ang, &s, &c);
  const size_t row = (size_t)(b * TT + t) * (3 * CC);
  const int hc = h * DD + dh;
  const float q1 = qkv[row + hc],          q2 = qkv[row + hc + 32];
  const float k1 = qkv[row + CC + hc],     k2 = qkv[row + CC + hc + 32];
  const float v1 = qkv[row + 2 * CC + hc], v2 = qkv[row + 2 * CC + hc + 32];
  const float qo1 = q1 * c - q2 * s, qo2 = q2 * c + q1 * s;
  const float ko1 = k1 * c - k2 * s, ko2 = k2 * c + k1 * s;
  const size_t bh = (size_t)(b * HH + h);
  // Q row-major
  bf16_t* qrow = Q + (bh * TT + t) * DD;
  qrow[dh]      = (bf16_t)qo1;
  qrow[dh + 32] = (bf16_t)qo2;
  // K packed: k-dim = d, n = t. kk = d%32 == dh for both halves.
  {
    const int nt = t >> 4;
    const int L  = ((dh >> 4) << 4) | (t & 15);
    const int e  = dh & 15;
    bf16_t* kb = Kp + bh * (size_t)(128 * 2 * 512);
    kb[((size_t)(nt * 2 + 0) << 9) + L * 16 + e] = (bf16_t)ko1;
    kb[((size_t)(nt * 2 + 1) << 9) + L * 16 + e] = (bf16_t)ko2;
  }
  // V packed: k-dim = t, n = d.
  {
    const int tc = t >> 5, kk = t & 31, e = t & 15;
    const int Lhi = (kk >> 4) << 4;
    bf16_t* vb = Vp + bh * (size_t)(64 * 4 * 512);
    const int n1 = dh >> 4,        L1 = Lhi | (dh & 15);
    const int n2 = (dh + 32) >> 4, L2 = Lhi | ((dh + 32) & 15);
    vb[((size_t)(tc * 4 + n1) << 9) + L1 * 16 + e] = (bf16_t)v1;
    vb[((size_t)(tc * 4 + n2) << 9) + L2 * 16 + e] = (bf16_t)v2;
  }
}

// ---------------------------------------------------------------------------
// Kernel 5: flash attention. One wave = 32 queries (2 M-tiles) of one (b,h);
// loop over 32-key blocks: S via 8 WMMA (K frags shared across M-tiles),
// online softmax, P->LDS transpose (D-layout -> A-layout), PV via 8 WMMA
// (V frags shared). Output written bf16 into attn matrix [B*T, C].
// ---------------------------------------------------------------------------
__global__ void __launch_bounds__(256)
attn_fa_wmma(const bf16_t* __restrict__ Q, const bf16_t* __restrict__ Kp,
             const bf16_t* __restrict__ Vp, bf16_t* __restrict__ Ob) {
  __shared__ __align__(16) bf16_t plds[8 * 1024];  // 2KB per wave (32x32 bf16)
  const int wave = blockIdx.x * 8 + (threadIdx.x >> 5);
  const int lane = threadIdx.x & 31;
  const int qt = wave & 63;                // 32-query tile
  const int h  = (wave >> 6) & 15;
  const int b  = wave >> 10;
  const size_t bh = (size_t)(b * HH + h);
  const int wslot = (threadIdx.x >> 5) * 1024;

  const bf16_t* qbase = Q + (bh * TT + (size_t)qt * 32) * DD;
  v16bf aq[2][2];
#pragma unroll
  for (int mt = 0; mt < 2; ++mt) {
    aq[mt][0] = load_a_frag_global(qbase + mt * 16 * DD, DD, 0, 0);
    aq[mt][1] = load_a_frag_global(qbase + mt * 16 * DD, DD, 0, 32);
  }
  const bf16_t* kb = Kp + bh * (size_t)(128 * 2 * 512);
  const bf16_t* vb = Vp + bh * (size_t)(64 * 4 * 512);

  v8f o[2][4] = {};
  float mrun[2][8], lrun[2][8];
#pragma unroll
  for (int mt = 0; mt < 2; ++mt)
#pragma unroll
    for (int r = 0; r < 8; ++r) { mrun[mt][r] = -1e30f; lrun[mt][r] = 0.0f; }
  const float scale = 0.125f;   // 1/sqrt(64)

  for (int kbk = 0; kbk < TT / 32; ++kbk) {
    // ---- scores for 32 key positions: K frags loaded once, used 2x ----
    v16bf bk00 = load_b_frag(kb + ((size_t)((2 * kbk + 0) * 2 + 0) << 9));
    v16bf bk01 = load_b_frag(kb + ((size_t)((2 * kbk + 0) * 2 + 1) << 9));
    v16bf bk10 = load_b_frag(kb + ((size_t)((2 * kbk + 1) * 2 + 0) << 9));
    v16bf bk11 = load_b_frag(kb + ((size_t)((2 * kbk + 1) * 2 + 1) << 9));
    v8f s[2][2] = {};
#pragma unroll
    for (int mt = 0; mt < 2; ++mt) {
      s[mt][0] = wmma_bf16(aq[mt][0], bk00, s[mt][0]);
      s[mt][0] = wmma_bf16(aq[mt][1], bk01, s[mt][0]);
      s[mt][1] = wmma_bf16(aq[mt][0], bk10, s[mt][1]);
      s[mt][1] = wmma_bf16(aq[mt][1], bk11, s[mt][1]);
    }

    // ---- online softmax (row = r + 8*(lane/16); cols across 16 lanes) ----
    const int mhalf = (lane >> 4) << 3;
    const int col = lane & 15;
#pragma unroll
    for (int mt = 0; mt < 2; ++mt) {
      float alpha[8];
#pragma unroll
      for (int r = 0; r < 8; ++r) {
        float a0 = s[mt][0][r] * scale, a1 = s[mt][1][r] * scale;
        float bm = fmaxf(a0, a1);
        bm = fmaxf(bm, __shfl_xor(bm, 1));
        bm = fmaxf(bm, __shfl_xor(bm, 2));
        bm = fmaxf(bm, __shfl_xor(bm, 4));
        bm = fmaxf(bm, __shfl_xor(bm, 8));
        const float mnew = fmaxf(mrun[mt][r], bm);
        const float p0 = __expf(a0 - mnew);
        const float p1 = __expf(a1 - mnew);
        float rs = p0 + p1;
        rs += __shfl_xor(rs, 1);
        rs += __shfl_xor(rs, 2);
        rs += __shfl_xor(rs, 4);
        rs += __shfl_xor(rs, 8);
        const float al = __expf(mrun[mt][r] - mnew);
        lrun[mt][r] = lrun[mt][r] * al + rs;
        mrun[mt][r] = mnew;
        alpha[r] = al;
        s[mt][0][r] = p0;
        s[mt][1][r] = p1;
      }
#pragma unroll
      for (int r = 0; r < 8; ++r)
#pragma unroll
        for (int j = 0; j < 4; ++j) o[mt][j][r] *= alpha[r];
      // ---- P (16x32) D-layout -> LDS row-major ----
#pragma unroll
      for (int r = 0; r < 8; ++r) {
        plds[wslot + (mt * 16 + mhalf + r) * 32 + col]      = (bf16_t)s[mt][0][r];
        plds[wslot + (mt * 16 + mhalf + r) * 32 + col + 16] = (bf16_t)s[mt][1][r];
      }
    }
    asm volatile("s_wait_dscnt 0" ::: "memory");   // store->load, same wave
    v16bf pa[2];
#pragma unroll
    for (int mt = 0; mt < 2; ++mt) {
      const int m = lane & 15;
      const int koff = (lane >> 4) << 4;
      v8bf lo = *(const v8bf*)&plds[wslot + (mt * 16 + m) * 32 + koff];
      v8bf hi = *(const v8bf*)&plds[wslot + (mt * 16 + m) * 32 + 8 + koff];
#pragma unroll
      for (int i = 0; i < 8; ++i) { pa[mt][i] = lo[i]; pa[mt][i + 8] = hi[i]; }
    }

    // ---- O += P @ V : V frags loaded once, used 2x ----
    const bf16_t* vkb = vb + ((size_t)(kbk * 4) << 9);
#pragma unroll
    for (int j = 0; j < 4; ++j) {
      v16bf bv = load_b_frag(vkb + j * 512);
      o[0][j] = wmma_bf16(pa[0], bv, o[0][j]);
      o[1][j] = wmma_bf16(pa[1], bv, o[1][j]);
    }
  }

  // ---- normalize and scatter bf16 into attn matrix [B*T, C] ----
  const int mhalf = (lane >> 4) << 3;
  const int col = lane & 15;
#pragma unroll
  for (int mt = 0; mt < 2; ++mt) {
#pragma unroll
    for (int r = 0; r < 8; ++r) {
      const float inv = 1.0f / lrun[mt][r];
      const size_t row = (size_t)(b * TT + qt * 32 + mt * 16 + mhalf + r);
      bf16_t* op = Ob + row * CC + h * DD + col;
#pragma unroll
      for (int j = 0; j < 4; ++j) op[j * 16] = (bf16_t)(o[mt][j][r] * inv);
    }
  }
}

// ---------------------------------------------------------------------------
extern "C" void kernel_launch(void* const* d_in, const int* in_sizes, int n_in,
                              void* d_out, int out_size, void* d_ws, size_t ws_size,
                              hipStream_t stream) {
  (void)in_sizes; (void)n_in; (void)out_size; (void)ws_size;
  const float* x    = (const float*)d_in[0];
  const float* Wqkv = (const float*)d_in[1];
  const float* bqkv = (const float*)d_in[2];
  const float* Wout = (const float*)d_in[3];
  const float* bout = (const float*)d_in[4];
  float* out = (float*)d_out;

  char* ws = (char*)d_ws;
  auto alloc = [&](size_t bytes) -> char* {
    char* p = ws;
    ws += (bytes + 255) & ~(size_t)255;
    return p;
  };
  bf16_t* xb    = (bf16_t*)alloc((size_t)4096 * 1024 * 2);       //  8.4 MB
  bf16_t* wqp   = (bf16_t*)alloc((size_t)1024 * 3072 * 2);       //  6.3 MB
  bf16_t* wop   = (bf16_t*)alloc((size_t)1024 * 1024 * 2);       //  2.1 MB
  float*  qkv   = (float*) alloc((size_t)4096 * 3072 * 4);       // 50.3 MB
  bf16_t* Qr    = (bf16_t*)alloc((size_t)32 * 2048 * 64 * 2);    //  8.4 MB
  bf16_t* Kpk   = (bf16_t*)alloc((size_t)32 * 2048 * 64 * 2);    //  8.4 MB
  bf16_t* Vpk   = (bf16_t*)alloc((size_t)32 * 2048 * 64 * 2);    //  8.4 MB
  bf16_t* attnb = (bf16_t*)alloc((size_t)4096 * 1024 * 2);       //  8.4 MB

  f32_to_bf16_v4<<<4096, 256, 0, stream>>>(x, xb);                    // 4.19M elems
  pack_b_wmma<<<12288, 256, 0, stream>>>(Wqkv, wqp, 1024, 3072);
  pack_b_wmma<<<4096, 256, 0, stream>>>(Wout, wop, 1024, 1024);
  // QKV GEMM: (4096/32)*(3072/64) = 6144 waves -> 768 blocks
  gemm_bf16_wmma<<<768, 256, 0, stream>>>(xb, wqp, bqkv, qkv, 4096, 3072, 1024);
  rope_pack<<<8192, 256, 0, stream>>>(qkv, Qr, Kpk, Vpk);
  // Attention: 2*16*(2048/32) = 2048 waves -> 256 blocks
  attn_fa_wmma<<<256, 256, 0, stream>>>(Qr, Kpk, Vpk, attnb);
  // Out GEMM: (4096/32)*(1024/64) = 2048 waves -> 256 blocks
  gemm_bf16_wmma<<<256, 256, 0, stream>>>(attnb, wop, bout, out, 4096, 1024, 1024);
}
